// GCN_base_91036126806369
// MI455X (gfx1250) — matrix operand order repaired and checked by reading, not compiled, not run on hardware
//
#include <hip/hip_runtime.h>
#include <hip/hip_bf16.h>

#define NFEAT 512
#define NHID  64
#define NCLASS 16

typedef __attribute__((ext_vector_type(16))) __bf16 v16bf;
typedef __attribute__((ext_vector_type(8)))  float v8f;
typedef __attribute__((ext_vector_type(2)))  float f2;

// ---------------------------------------------------------------------------
// Weight pre-swizzle into WMMA B-fragment order (bf16).
// Fragment linear layout: ((kt*NT + nt)*32 + lane)*16 + j
//   lane 0-15 : N = nt*16 + lane,      K = kt*32 + j
//   lane 16-31: N = nt*16 + (lane-16), K = kt*32 + 16 + j
// ---------------------------------------------------------------------------
__global__ void k_convW1(const float* __restrict__ W1, __bf16* __restrict__ W1b) {
    int idx = blockIdx.x * blockDim.x + threadIdx.x;
    if (idx >= NFEAT * NHID) return;
    int j    = idx & 15;
    int lane = (idx >> 4) & 31;
    int t    = idx >> 9;           // kt*4 + nt
    int nt   = t & 3;
    int kt   = t >> 2;
    int n = nt * 16 + (lane & 15);
    int k = kt * 32 + ((lane & 16) ? 16 : 0) + j;
    W1b[idx] = (__bf16)W1[k * NHID + n];
}

__global__ void k_convW2(const float* __restrict__ W2, __bf16* __restrict__ W2b) {
    int idx = blockIdx.x * blockDim.x + threadIdx.x;
    if (idx >= NHID * NCLASS) return;
    int j    = idx & 15;
    int lane = (idx >> 4) & 31;
    int kt   = idx >> 9;           // 0..1
    int n = lane & 15;
    int k = kt * 32 + ((lane & 16) ? 16 : 0) + j;
    W2b[idx] = (__bf16)W2[k * NCLASS + n];
}

// ---------------------------------------------------------------------------
// Degree / normalization
// ---------------------------------------------------------------------------
__global__ void k_deg_init(float* deg, int N) {
    int i = blockIdx.x * blockDim.x + threadIdx.x;
    if (i < N) deg[i] = 1.0f;                       // self loop
}
__global__ void k_deg_count(const int* __restrict__ dst, float* deg, int E) {
    int e = blockIdx.x * blockDim.x + threadIdx.x;
    if (e < E) unsafeAtomicAdd(&deg[dst[e]], 1.0f);
}
__global__ void k_deg_rsqrt(float* deg, int N) {
    int i = blockIdx.x * blockDim.x + threadIdx.x;
    if (i < N) deg[i] = rsqrtf(fmaxf(deg[i], 1.0f));
}

// ---------------------------------------------------------------------------
// GEMM1: h1[N,64] = x[N,512] @ W1  (bf16 WMMA, f32 accumulate)
// 4 waves / block; each wave computes 32 rows x 64 cols (2 A-frags x 4 B-frags
// = 8 WMMAs per k-step, B fragments reused twice). W1 bf16 fragments in LDS.
// ---------------------------------------------------------------------------
__global__ void __launch_bounds__(128) k_gemm1(const float* __restrict__ x,
                                               const __bf16* __restrict__ W1b,
                                               float* __restrict__ h1, int N) {
    __shared__ __align__(32) __bf16 sW[NFEAT * NHID];   // 64 KB
    {
        const uint4* srcp = (const uint4*)W1b;
        uint4* dstp = (uint4*)sW;
        for (int i = threadIdx.x; i < (NFEAT * NHID) / 8; i += blockDim.x)
            dstp[i] = srcp[i];
    }
    __syncthreads();

    int wave = threadIdx.x >> 5;
    int lane = threadIdx.x & 31;
    int m0 = (blockIdx.x * 4 + wave) * 32;
    if (m0 >= N) return;
    int half = (lane >> 4) & 1;
    const float* xr0 = x + (size_t)(m0 + (lane & 15)) * NFEAT;        // rows m0..m0+15
    const float* xr1 = x + (size_t)(m0 + 16 + (lane & 15)) * NFEAT;   // rows m0+16..m0+31

    v8f acc[2][4] = {};
    const int koffs[8] = {0, 2, 4, 6, 8, 10, 12, 14};

    for (int kt = 0; kt < NFEAT / 32; ++kt) {
        int k0 = kt * 32 + half * 16;
        if (kt < NFEAT / 32 - 1) {                  // software prefetch next chunk
            __builtin_prefetch(xr0 + k0 + 32, 0, 1);
            __builtin_prefetch(xr1 + k0 + 32, 0, 1);
        }
        v16bf a0, a1;
#pragma unroll
        for (int i = 0; i < 8; ++i) {
            f2 v0 = *(const f2*)(xr0 + k0 + koffs[i]);
            f2 v1 = *(const f2*)(xr1 + k0 + koffs[i]);
            a0[2 * i]     = (__bf16)v0.x;
            a0[2 * i + 1] = (__bf16)v0.y;
            a1[2 * i]     = (__bf16)v1.x;
            a1[2 * i + 1] = (__bf16)v1.y;
        }
#pragma unroll
        for (int nt = 0; nt < 4; ++nt) {
            v16bf b = *(const v16bf*)(sW + (((kt * 4 + nt) * 32 + lane) << 4));
            acc[0][nt] = __builtin_amdgcn_wmma_f32_16x16x32_bf16(false, a0, false, b,
                                                                 (short)0, acc[0][nt], false, false);
            acc[1][nt] = __builtin_amdgcn_wmma_f32_16x16x32_bf16(false, a1, false, b,
                                                                 (short)0, acc[1][nt], false, false);
        }
    }

#pragma unroll
    for (int mt = 0; mt < 2; ++mt) {
#pragma unroll
        for (int r = 0; r < 8; ++r) {
            int row = m0 + mt * 16 + r + 8 * half;
            float* o = h1 + (size_t)row * NHID + (lane & 15);
            o[0]  = acc[mt][0][r];
            o[16] = acc[mt][1][r];
            o[32] = acc[mt][2][r];
            o[48] = acc[mt][3][r];
        }
    }
}

// ---------------------------------------------------------------------------
// GEMM2: h2[N,16] = relu(h1_agg)[N,64] @ W2  (bf16 WMMA, fused ReLU)
// ---------------------------------------------------------------------------
__global__ void __launch_bounds__(128) k_gemm2(const float* __restrict__ h1a,
                                               const __bf16* __restrict__ W2b,
                                               float* __restrict__ h2, int N) {
    int wave = threadIdx.x >> 5;
    int lane = threadIdx.x & 31;
    int m0 = (blockIdx.x * 4 + wave) * 16;
    if (m0 >= N) return;
    int half = (lane >> 4) & 1;
    const float* hr = h1a + (size_t)(m0 + (lane & 15)) * NHID;

    v16bf bf0 = *(const v16bf*)(W2b + ((0 * 32 + lane) << 4));
    v16bf bf1 = *(const v16bf*)(W2b + ((1 * 32 + lane) << 4));

    const int koffs[8] = {0, 2, 4, 6, 8, 10, 12, 14};
    v8f acc = {};
#pragma unroll
    for (int kt = 0; kt < 2; ++kt) {
        int k0 = kt * 32 + half * 16;
        v16bf a;
#pragma unroll
        for (int i = 0; i < 8; ++i) {
            f2 v = *(const f2*)(hr + k0 + koffs[i]);
            a[2 * i]     = (__bf16)fmaxf(v.x, 0.0f);   // fused ReLU
            a[2 * i + 1] = (__bf16)fmaxf(v.y, 0.0f);
        }
        acc = __builtin_amdgcn_wmma_f32_16x16x32_bf16(false, a, false, (kt == 0) ? bf0 : bf1,
                                                      (short)0, acc, false, false);
    }
#pragma unroll
    for (int r = 0; r < 8; ++r)
        h2[(size_t)(m0 + r + 8 * half) * NCLASS + (lane & 15)] = acc[r];
}

// ---------------------------------------------------------------------------
// Aggregation: init with bias + self-loop, then edge scatter-add
// ---------------------------------------------------------------------------
__global__ void k_agg_init(const float* __restrict__ h, const float* __restrict__ bias,
                           const float* __restrict__ dis, float* __restrict__ ha,
                           int N, int F) {
    int idx = blockIdx.x * blockDim.x + threadIdx.x;
    if (idx >= N * F) return;
    int n = idx / F;
    int f = idx - n * F;
    float d = dis[n];
    ha[idx] = bias[f] + h[idx] * d * d;
}

__global__ void k_agg_edges64(const int* __restrict__ src, const int* __restrict__ dst,
                              const float* __restrict__ dis,
                              const float* __restrict__ h, float* __restrict__ ha, int E) {
    int gid = blockIdx.x * blockDim.x + threadIdx.x;
    int e = gid >> 6;
    if (e >= E) return;
    int f = gid & 63;
    int s = src[e], d = dst[e];
    float norm = dis[s] * dis[d];
    unsafeAtomicAdd(&ha[(size_t)d * 64 + f], h[(size_t)s * 64 + f] * norm);
}

__global__ void k_agg_edges16(const int* __restrict__ src, const int* __restrict__ dst,
                              const float* __restrict__ dis,
                              const float* __restrict__ h, float* __restrict__ ha, int E) {
    int gid = blockIdx.x * blockDim.x + threadIdx.x;
    int e = gid >> 4;
    if (e >= E) return;
    int f = gid & 15;
    int s = src[e], d = dst[e];
    float norm = dis[s] * dis[d];
    unsafeAtomicAdd(&ha[(size_t)d * 16 + f], h[(size_t)s * 16 + f] * norm);
}

// ---------------------------------------------------------------------------
// Mean pool + log_softmax
// ---------------------------------------------------------------------------
__global__ void k_pool_zero(float* pool) {
    if (threadIdx.x < NCLASS) pool[threadIdx.x] = 0.0f;
}

__global__ void k_pool_reduce(const float* __restrict__ h2a, float* __restrict__ pool, int N) {
    __shared__ float s[NCLASS];
    if (threadIdx.x < NCLASS) s[threadIdx.x] = 0.0f;
    __syncthreads();
    int n = blockIdx.x * 16 + (threadIdx.x >> 4);
    int f = threadIdx.x & 15;
    if (n < N) atomicAdd(&s[f], h2a[(size_t)n * NCLASS + f]);  // ds_add_f32
    __syncthreads();
    if (threadIdx.x < NCLASS) unsafeAtomicAdd(&pool[threadIdx.x], s[threadIdx.x]);
}

__global__ void k_final(const float* __restrict__ pool, float* __restrict__ out, int N) {
    if (threadIdx.x != 0 || blockIdx.x != 0) return;
    float g[NCLASS];
    float mx = -3.0e38f;
    for (int f = 0; f < NCLASS; ++f) {
        g[f] = fmaxf(pool[f] / (float)N, 0.0f);
        mx = fmaxf(mx, g[f]);
    }
    float ssum = 0.0f;
    for (int f = 0; f < NCLASS; ++f) ssum += __expf(g[f] - mx);
    float l = __logf(ssum);
    for (int f = 0; f < NCLASS; ++f) out[f] = g[f] - mx - l;
}

// ---------------------------------------------------------------------------
extern "C" void kernel_launch(void* const* d_in, const int* in_sizes, int n_in,
                              void* d_out, int out_size, void* d_ws, size_t ws_size,
                              hipStream_t stream) {
    const float* x  = (const float*)d_in[0];
    const int*   ei = (const int*)d_in[1];
    const float* W1 = (const float*)d_in[2];
    const float* b1 = (const float*)d_in[3];
    const float* W2 = (const float*)d_in[4];
    const float* b2 = (const float*)d_in[5];
    float* out = (float*)d_out;

    const int N = in_sizes[0] / NFEAT;
    const int E = in_sizes[1] / 2;
    const int* src = ei;
    const int* dst = ei + E;

    // workspace layout
    char* ws = (char*)d_ws;
    size_t off = 0;
    auto take = [&](size_t bytes) -> char* {
        char* p = ws + off;
        off = (off + bytes + 255) & ~(size_t)255;
        return p;
    };
    float*  dis = (float*)take((size_t)N * 4);
    __bf16* W1b = (__bf16*)take((size_t)NFEAT * NHID * 2);
    __bf16* W2b = (__bf16*)take((size_t)NHID * NCLASS * 2);
    float*  h1  = (float*)take((size_t)N * NHID * 4);
    float*  h1a = (float*)take((size_t)N * NHID * 4);
    float*  h2  = (float*)take((size_t)N * NCLASS * 4);
    float*  h2a = (float*)take((size_t)N * NCLASS * 4);
    float*  pool = (float*)take(64);
    (void)ws_size;

    const int B = 256;
    // weights -> bf16 fragment images
    k_convW1<<<(NFEAT * NHID + B - 1) / B, B, 0, stream>>>(W1, W1b);
    k_convW2<<<(NHID * NCLASS + B - 1) / B, B, 0, stream>>>(W2, W2b);

    // degree / normalization
    k_deg_init<<<(N + B - 1) / B, B, 0, stream>>>(dis, N);
    k_deg_count<<<(E + B - 1) / B, B, 0, stream>>>(dst, dis, E);
    k_deg_rsqrt<<<(N + B - 1) / B, B, 0, stream>>>(dis, N);

    // layer 1
    int tiles32 = (N + 31) / 32;
    int g1Blocks = (tiles32 + 3) / 4;
    k_gemm1<<<g1Blocks, 128, 0, stream>>>(x, W1b, h1, N);
    k_agg_init<<<((size_t)N * NHID + B - 1) / B, B, 0, stream>>>(h1, b1, dis, h1a, N, NHID);
    k_agg_edges64<<<((size_t)E * 64 + B - 1) / B, B, 0, stream>>>(src, dst, dis, h1, h1a, E);

    // layer 2 (ReLU fused into GEMM2 A-load)
    int tiles16 = (N + 15) / 16;
    int g2Blocks = (tiles16 + 3) / 4;
    k_gemm2<<<g2Blocks, 128, 0, stream>>>(h1a, W2b, h2, N);
    k_agg_init<<<((size_t)N * NCLASS + B - 1) / B, B, 0, stream>>>(h2, b2, dis, h2a, N, NCLASS);
    k_agg_edges16<<<((size_t)E * 16 + B - 1) / B, B, 0, stream>>>(src, dst, dis, h2, h2a, E);

    // pool + log_softmax
    k_pool_zero<<<1, 32, 0, stream>>>(pool);
    k_pool_reduce<<<(N + 15) / 16, 256, 0, stream>>>(h2a, pool, N);
    k_final<<<1, 32, 0, stream>>>(pool, out, N);
}
